// BiLSTM_CRF_36043365548819
// MI455X (gfx1250) — matrix operand order repaired and checked by reading, not compiled, not run on hardware
//
#include <hip/hip_runtime.h>
#include <hip/hip_bf16.h>

// BiLSTM-CRF forward for MI455X (gfx1250, wave32, WMMA f32_16x16x32_f16).
//
// Pipeline:
//   packB      : repack W_ih_f/b, W_hh_f/b, W_out(+pad to 32) into per-lane
//                WMMA B-fragment layout (f16).
//   embed_gate : xg[dir][l*B+b][4H] = f16( emb[x[b,l]] @ W_ih^T + b_ih + b_hh )
//   lstm_rec   : 2 persistent workgroups (one per direction), 512 sequential
//                steps, h@W_hh^T via WMMA, gates via VALU, h/c state in LDS/regs.
//                xg[t] prefetched (global_prefetch_b8) during the WMMA phase.
//   outproj    : out = concat(hs_f,hs_b) @ W_out^T + b_out, then CRF broadcast
//                d_out[b,l,i,j] = out[b,l,j] + transition[i,j].

typedef __attribute__((ext_vector_type(16))) _Float16 v16h;
typedef __attribute__((ext_vector_type(8)))  _Float16 v8h;
typedef __attribute__((ext_vector_type(8)))  float    v8f;

#define WMMA16(a,b,c) __builtin_amdgcn_wmma_f32_16x16x32_f16(false,(a),false,(b),(short)0,(c),false,false)

__device__ __forceinline__ v8f vzero8() {
    v8f z = {0.f,0.f,0.f,0.f,0.f,0.f,0.f,0.f};
    return z;
}

// A-fragment (16x32 f16, per-lane: halves 0..7 -> k0..k0+7, halves 8..15 -> k0+16..k0+23)
// loaded from an f32 row (emb).
__device__ __forceinline__ v16h afrag_f32(const float* __restrict__ row, int k0) {
    const float4* p = (const float4*)(row + k0);
    const float4* q = (const float4*)(row + k0 + 16);
    float4 a0 = p[0], a1 = p[1], b0 = q[0], b1 = q[1];
    union { v16h v; _Float16 e[16]; } u;
    u.e[0]=(_Float16)a0.x; u.e[1]=(_Float16)a0.y; u.e[2]=(_Float16)a0.z; u.e[3]=(_Float16)a0.w;
    u.e[4]=(_Float16)a1.x; u.e[5]=(_Float16)a1.y; u.e[6]=(_Float16)a1.z; u.e[7]=(_Float16)a1.w;
    u.e[8]=(_Float16)b0.x; u.e[9]=(_Float16)b0.y; u.e[10]=(_Float16)b0.z; u.e[11]=(_Float16)b0.w;
    u.e[12]=(_Float16)b1.x; u.e[13]=(_Float16)b1.y; u.e[14]=(_Float16)b1.z; u.e[15]=(_Float16)b1.w;
    return u.v;
}

// Same A-fragment loaded from an f16 row (hs).
__device__ __forceinline__ v16h afrag_f16(const _Float16* __restrict__ row, int k0) {
    union { v16h v; v8h h[2]; } u;
    u.h[0] = *(const v8h*)(row + k0);
    u.h[1] = *(const v8h*)(row + k0 + 16);
    return u.v;
}

// -------------------------------------------------------------------------
// packB: dst fragment layout, one f16 per output element.
//   element i: j = i&15 (half within lane), lane = (i>>4)&31,
//              kc = (i>>9)%kchunks, nt = (i>>9)/kchunks
//   B[k][n] = W[n][k] (we multiply by W^T), n = nt*16 + lane%16,
//   k = kc*32 + (lane/16)*8 + (j&7) + (j/8)*16  (B-matrix 32x16 16-bit per-lane layout)
// -------------------------------------------------------------------------
__global__ void packB(const float* __restrict__ src, _Float16* __restrict__ dst,
                      int Nrows, int K, int kchunks, int total) {
    int i = blockIdx.x * 256 + threadIdx.x;
    if (i >= total) return;
    int j    = i & 15;
    int lane = (i >> 4) & 31;
    int f    = i >> 9;
    int kc   = f % kchunks;
    int nt   = f / kchunks;
    int n = nt * 16 + (lane & 15);
    int k = kc * 32 + (lane >> 4) * 8 + (j & 7) + (j >> 3) * 16;
    float v = (n < Nrows) ? src[(size_t)n * K + k] : 0.f;
    dst[i] = (_Float16)v;
}

// -------------------------------------------------------------------------
// embed_gate: one wave per (mtile, ntile-pair, dir). M = L*B (row = l*64+b),
// N = 4H = 1024 (64 ntiles/dir), K = E = 128 (4 kchunks).
// -------------------------------------------------------------------------
__global__ void __launch_bounds__(256) embed_gate(
    const int* __restrict__ x, const float* __restrict__ emb,
    const _Float16* __restrict__ pWih0, const _Float16* __restrict__ pWih1,
    const float* __restrict__ bih0, const float* __restrict__ bhh0,
    const float* __restrict__ bih1, const float* __restrict__ bhh1,
    _Float16* __restrict__ xg0, _Float16* __restrict__ xg1) {
    int lane = threadIdx.x & 31;
    int wid  = (blockIdx.x * 256 + threadIdx.x) >> 5;   // 0 .. 131071
    int mtile = wid >> 6;                               // 0 .. 2047
    int p     = wid & 63;                               // ntile-pair task
    int dir   = p >> 5;
    int ntl   = (p & 31) * 2;                           // local ntile (of 64)
    const _Float16* pW = dir ? pWih1 : pWih0;
    const float* bi = dir ? bih1 : bih0;
    const float* bh = dir ? bhh1 : bhh0;
    _Float16* xg = dir ? xg1 : xg0;

    int m = mtile * 16 + (lane & 15);
    int l = m >> 6, b = m & 63;
    int tok = x[b * 512 + l];
    const float* erow = emb + (size_t)tok * 128;

    v8f acc0 = vzero8(), acc1 = vzero8();
#pragma unroll
    for (int kc = 0; kc < 4; kc++) {
        v16h a  = afrag_f32(erow, kc * 32 + (lane >> 4) * 8);
        v16h b0 = *(const v16h*)(pW + ((size_t)(ntl * 4 + kc) * 512 + lane * 16));
        v16h b1 = *(const v16h*)(pW + ((size_t)((ntl + 1) * 4 + kc) * 512 + lane * 16));
        acc0 = WMMA16(a, b0, acc0);
        acc1 = WMMA16(a, b1, acc1);
    }
    // C/D layout: VGPR v -> row m_local = v + 8*(lane>=16), col = lane%16
#pragma unroll
    for (int nt2 = 0; nt2 < 2; nt2++) {
        int col = (ntl + nt2) * 16 + (lane & 15);
        float bias = bi[col] + bh[col];
        v8f a = nt2 ? acc1 : acc0;
#pragma unroll
        for (int v = 0; v < 8; v++) {
            int row = mtile * 16 + v + ((lane >> 4) << 3);
            xg[(size_t)row * 1024 + col] = (_Float16)(a[v] + bias);
        }
    }
}

// -------------------------------------------------------------------------
// lstm_rec: grid = 2 blocks (dir 0 forward, dir 1 backward), 1024 threads.
// LDS: gmat f32[64][1024] (256KB) + hsh f16[64][256] (32KB) = 288KB dynamic.
// Wave w owns ntiles {2w, 2w+1}; loops mtiles in 2 half-passes to cap VGPRs.
// Cell state c lives in registers (16 floats/thread). xg[t] (128KB/step) is
// prefetched into cache at the start of the WMMA phase so the gate phase
// doesn't stall on L2/HBM latency.
// -------------------------------------------------------------------------
__global__ void __launch_bounds__(1024, 1) lstm_rec(
    const _Float16* __restrict__ xg0, const _Float16* __restrict__ xg1,
    const _Float16* __restrict__ pW0, const _Float16* __restrict__ pW1,
    _Float16* __restrict__ hs0, _Float16* __restrict__ hs1) {
    extern __shared__ char smem[];
    float*    gmat = (float*)smem;                      // [64][1024]
    _Float16* hsh  = (_Float16*)(smem + 64 * 1024 * 4); // [64][256]

    const int dir = blockIdx.x;
    const _Float16* xg = dir ? xg1 : xg0;
    const _Float16* pW = dir ? pW1 : pW0;
    _Float16* hs = dir ? hs1 : hs0;

    int tid = threadIdx.x, lane = tid & 31, w = tid >> 5;
    int nt0 = w * 2;                                   // ntiles nt0, nt0+1 of 64
    int gb = tid >> 4;                                 // batch row for gate phase
    int j0 = tid & 15;                                 // hidden base (j = j0 + 16q)

    for (int i = tid; i < 64 * 256; i += 1024) hsh[i] = (_Float16)0.f;
    float c[16];
#pragma unroll
    for (int q = 0; q < 16; q++) c[q] = 0.f;
    __syncthreads();

    for (int s = 0; s < 512; s++) {
        int t = dir ? (511 - s) : s;

        // Prefetch this step's xg row slice (row gb is 2KB; 16 threads per row
        // each cover one 128B line) so it lands in cache while WMMAs execute.
        {
            const char* pfx = (const char*)(xg + ((size_t)(t * 64 + gb)) * 1024)
                              + (size_t)(tid & 15) * 128;
            __builtin_prefetch(pfx, 0, 3);
        }

        // ---- matmul phase: G = h @ W_hh^T (f16 WMMA, f32 accum) ----
#pragma unroll
        for (int mth = 0; mth < 2; mth++) {
            v8f acc[2][2] = {{vzero8(), vzero8()}, {vzero8(), vzero8()}};
            for (int kc = 0; kc < 8; kc++) {
                v16h b0 = *(const v16h*)(pW + ((size_t)(nt0 * 8 + kc) * 512 + lane * 16));
                v16h b1 = *(const v16h*)(pW + ((size_t)((nt0 + 1) * 8 + kc) * 512 + lane * 16));
#pragma unroll
                for (int m2 = 0; m2 < 2; m2++) {
                    int mt = mth * 2 + m2;
                    const _Float16* ap = hsh + (mt * 16 + (lane & 15)) * 256
                                             + kc * 32 + (lane >> 4) * 8;
                    union { v16h v; v8h h[2]; } ua;
                    ua.h[0] = *(const v8h*)ap;
                    ua.h[1] = *(const v8h*)(ap + 16);
                    acc[m2][0] = WMMA16(ua.v, b0, acc[m2][0]);
                    acc[m2][1] = WMMA16(ua.v, b1, acc[m2][1]);
                }
            }
#pragma unroll
            for (int m2 = 0; m2 < 2; m2++) {
                int rbase = (mth * 2 + m2) * 16 + ((lane >> 4) << 3);
#pragma unroll
                for (int nt2 = 0; nt2 < 2; nt2++) {
                    int col = (nt0 + nt2) * 16 + (lane & 15);
#pragma unroll
                    for (int v = 0; v < 8; v++)
                        gmat[(rbase + v) * 1024 + col] = acc[m2][nt2][v];
                }
            }
        }
        __syncthreads();
        // ---- gate phase: i,f,g,o -> c,h ----
        {
            const _Float16* xr = xg + ((size_t)(t * 64 + gb)) * 1024;
            _Float16* hr = hs + ((size_t)(t * 64 + gb)) * 256;
#pragma unroll
            for (int q = 0; q < 16; q++) {
                int j = j0 + q * 16;
                float gi = gmat[gb * 1024 + j]        + (float)xr[j];
                float gf = gmat[gb * 1024 + 256 + j]  + (float)xr[256 + j];
                float gg = gmat[gb * 1024 + 512 + j]  + (float)xr[512 + j];
                float go = gmat[gb * 1024 + 768 + j]  + (float)xr[768 + j];
                float si = 1.f / (1.f + __expf(-gi));
                float sf = 1.f / (1.f + __expf(-gf));
                float so = 1.f / (1.f + __expf(-go));
                float cn = sf * c[q] + si * tanhf(gg);
                c[q] = cn;
                float h = so * tanhf(cn);
                hsh[gb * 256 + j] = (_Float16)h;
                hr[j] = (_Float16)h;
            }
        }
        __syncthreads();
    }
}

// -------------------------------------------------------------------------
// outproj: out[m][n] = concat(hs_f,hs_b)[m] . W_out[n] + b_out[n], then CRF
// broadcast. M = 32768 (row = l*64+b), N = 32 (24 valid), K = 512 (16 kchunks).
// One wave per M-tile; 8 waves/block; 256 blocks.
// -------------------------------------------------------------------------
__global__ void __launch_bounds__(256) outproj(
    const _Float16* __restrict__ hsf, const _Float16* __restrict__ hsb,
    const _Float16* __restrict__ pWout, const float* __restrict__ bout,
    const float* __restrict__ trans, float* __restrict__ out) {
    __shared__ float strans[576];
    __shared__ float sbout[24];
    for (int i = threadIdx.x; i < 576; i += 256) strans[i] = trans[i];
    if (threadIdx.x < 24) sbout[threadIdx.x] = bout[threadIdx.x];
    __syncthreads();

    int lane = threadIdx.x & 31, w = threadIdx.x >> 5;
    int mtile = blockIdx.x * 8 + w;                    // 0..2047
    int mrow = mtile * 16 + (lane & 15);

    v8f acc0 = vzero8(), acc1 = vzero8();
#pragma unroll
    for (int kc = 0; kc < 16; kc++) {
        const _Float16* row = (kc < 8) ? (hsf + (size_t)mrow * 256)
                                       : (hsb + (size_t)mrow * 256);
        int k0 = (kc & 7) * 32 + (lane >> 4) * 8;
        v16h a  = afrag_f16(row, k0);
        v16h b0 = *(const v16h*)(pWout + ((size_t)(0 * 16 + kc) * 512 + lane * 16));
        v16h b1 = *(const v16h*)(pWout + ((size_t)(1 * 16 + kc) * 512 + lane * 16));
        acc0 = WMMA16(a, b0, acc0);
        acc1 = WMMA16(a, b1, acc1);
    }
#pragma unroll
    for (int nt = 0; nt < 2; nt++) {
        int n = nt * 16 + (lane & 15);
        if (n < 24) {
            v8f a = nt ? acc1 : acc0;
            float bb = sbout[n];
#pragma unroll
            for (int v = 0; v < 8; v++) {
                int m = mtile * 16 + v + ((lane >> 4) << 3);
                int l = m >> 6, b = m & 63;
                float val = a[v] + bb;
                size_t base = ((size_t)b * 512 + l) * 576 + n;   // [b][l][i][n]
                for (int i = 0; i < 24; i++)
                    out[base + (size_t)i * 24] = val + strans[i * 24 + n];
            }
        }
    }
}

// -------------------------------------------------------------------------
// Host
// -------------------------------------------------------------------------
extern "C" void kernel_launch(void* const* d_in, const int* in_sizes, int n_in,
                              void* d_out, int out_size, void* d_ws, size_t ws_size,
                              hipStream_t stream) {
    const int*   x      = (const int*)  d_in[0];
    const float* emb    = (const float*)d_in[1];
    const float* W_ih_f = (const float*)d_in[2];
    const float* W_hh_f = (const float*)d_in[3];
    const float* b_ih_f = (const float*)d_in[4];
    const float* b_hh_f = (const float*)d_in[5];
    const float* W_ih_b = (const float*)d_in[6];
    const float* W_hh_b = (const float*)d_in[7];
    const float* b_ih_b = (const float*)d_in[8];
    const float* b_hh_b = (const float*)d_in[9];
    const float* W_out  = (const float*)d_in[10];
    const float* b_out  = (const float*)d_in[11];
    const float* transition = (const float*)d_in[12];
    float* out = (float*)d_out;

    char* ws = (char*)d_ws;
    const size_t OFF_PWIH0 = 0;                         // 64nt*4kc*512 halves = 256KB
    const size_t OFF_PWIH1 = 262144;
    const size_t OFF_PWHH0 = 524288;                    // 64nt*8kc*512 halves = 512KB
    const size_t OFF_PWHH1 = 1048576;
    const size_t OFF_PWOUT = 1572864;                   // 2nt*16kc*512 halves = 32KB
    const size_t OFF_XG0   = 1605632;                   // 512*64*1024 halves = 64MB
    const size_t OFF_XG1   = OFF_XG0 + 67108864ull;
    const size_t OFF_HS0   = OFF_XG1 + 67108864ull;     // 512*64*256 halves = 16MB
    const size_t OFF_HS1   = OFF_HS0 + 16777216ull;

    _Float16* pWih0 = (_Float16*)(ws + OFF_PWIH0);
    _Float16* pWih1 = (_Float16*)(ws + OFF_PWIH1);
    _Float16* pWhh0 = (_Float16*)(ws + OFF_PWHH0);
    _Float16* pWhh1 = (_Float16*)(ws + OFF_PWHH1);
    _Float16* pWout = (_Float16*)(ws + OFF_PWOUT);
    _Float16* xg0   = (_Float16*)(ws + OFF_XG0);
    _Float16* xg1   = (_Float16*)(ws + OFF_XG1);
    _Float16* hs0   = (_Float16*)(ws + OFF_HS0);
    _Float16* hs1   = (_Float16*)(ws + OFF_HS1);

    // 1) pack weights into WMMA B-fragment layout
    packB<<<(131072 + 255) / 256, 256, 0, stream>>>(W_ih_f, pWih0, 1024, 128, 4, 131072);
    packB<<<(131072 + 255) / 256, 256, 0, stream>>>(W_ih_b, pWih1, 1024, 128, 4, 131072);
    packB<<<(262144 + 255) / 256, 256, 0, stream>>>(W_hh_f, pWhh0, 1024, 256, 8, 262144);
    packB<<<(262144 + 255) / 256, 256, 0, stream>>>(W_hh_b, pWhh1, 1024, 256, 8, 262144);
    packB<<<(16384  + 255) / 256, 256, 0, stream>>>(W_out,  pWout, 24,   512, 16, 16384);

    // 2) embedding + input-gate GEMM (biases folded)
    embed_gate<<<16384, 256, 0, stream>>>(x, emb, pWih0, pWih1,
                                          b_ih_f, b_hh_f, b_ih_b, b_hh_b,
                                          xg0, xg1);

    // 3) bidirectional LSTM recurrence (288KB dynamic LDS per workgroup)
    (void)hipFuncSetAttribute((const void*)lstm_rec,
                              hipFuncAttributeMaxDynamicSharedMemorySize, 294912);
    lstm_rec<<<2, 1024, 294912, stream>>>(xg0, xg1, pWhh0, pWhh1, hs0, hs1);

    // 4) output projection + CRF broadcast
    outproj<<<256, 256, 0, stream>>>(hs0, hs1, pWout, b_out, transition, out);
}
